// OrkyHyenaModel_19258633356061
// MI455X (gfx1250) — matrix operand order repaired and checked by reading, not compiled, not run on hardware
//
#include <hip/hip_runtime.h>
#include <hip/hip_bf16.h>

// ---------------- problem constants ----------------
static constexpr int V  = 32000;
static constexpr int D  = 1024;
static constexpr int NL = 6;
static constexpr int S  = 2048;
static constexpr int B  = 2;
static constexpr int BS = B * S;              // 4096 rows
static constexpr long long BSD = (long long)BS * D;
static constexpr int KT = D / 32;             // K-tiles per GEMM (K is always D)

typedef __attribute__((ext_vector_type(16))) __bf16 v16bf;
typedef __attribute__((ext_vector_type(8)))  float  v8f;

union FragA { v16bf v; unsigned int u[8]; unsigned short h[16]; };
union FragB { v16bf v; unsigned int u[8]; unsigned short h[16]; };

__device__ __forceinline__ unsigned short f2bf(float f) {
    union { float f; unsigned u; } c; c.f = f;
    unsigned r = c.u + 0x7FFFu + ((c.u >> 16) & 1u);   // round-to-nearest-even
    return (unsigned short)(r >> 16);
}

#if __has_builtin(__builtin_amdgcn_global_load_async_to_lds_b32)
#define HAS_ASYNC_LDS 1
typedef __attribute__((address_space(1))) int* gas_ptr;
typedef __attribute__((address_space(3))) int* las_ptr;
#endif

__device__ __forceinline__ void wait_async_all() {
#if __has_builtin(__builtin_amdgcn_s_wait_asynccnt)
    __builtin_amdgcn_s_wait_asynccnt(0);
#else
    asm volatile("s_wait_asynccnt 0" ::: "memory");
#endif
}

// ---------------- 1. embedding + sinusoid ----------------
__global__ void embed_kernel(const int* __restrict__ ids,
                             const float* __restrict__ emb,
                             float* __restrict__ x) {
    int row = blockIdx.x;                 // b*S + s
    int s   = row % S;
    int id  = ids[row];
    const float c0 = -logf(10000.0f) / (float)D;
    for (int d = threadIdx.x; d < D; d += blockDim.x) {
        int i = d >> 1;
        float f   = __expf(c0 * (float)(2 * i));
        float ang = (float)s * f;
        float pe  = (d & 1) ? __cosf(ang) : __sinf(ang);
        x[(long long)row * D + d] = emb[(long long)id * D + d] + pe;
    }
}

// ---------------- 2. layernorm (block per row) ----------------
__global__ void ln_kernel(const float* __restrict__ x,
                          const float* __restrict__ g,
                          const float* __restrict__ b,
                          float* __restrict__ out) {
    __shared__ float sSum[256];
    __shared__ float sSq[256];
    int row = blockIdx.x;
    const float* xr = x + (long long)row * D;
    float s = 0.f, s2 = 0.f;
    for (int d = threadIdx.x; d < D; d += 256) { float v = xr[d]; s += v; s2 += v * v; }
    sSum[threadIdx.x] = s; sSq[threadIdx.x] = s2;
    __syncthreads();
    for (int off = 128; off > 0; off >>= 1) {
        if (threadIdx.x < off) {
            sSum[threadIdx.x] += sSum[threadIdx.x + off];
            sSq[threadIdx.x]  += sSq[threadIdx.x + off];
        }
        __syncthreads();
    }
    float mu  = sSum[0] * (1.0f / D);
    float var = sSq[0] * (1.0f / D) - mu * mu;
    float inv = rsqrtf(var + 1e-5f);
    float* o = out + (long long)row * D;
    for (int d = threadIdx.x; d < D; d += 256)
        o[d] = (xr[d] - mu) * inv * g[d] + b[d];
}

// ---------------- 3. depthwise conv k=3, pad (1,1) ----------------
__global__ void dwconv3_kernel(const float* __restrict__ h,
                               const float* __restrict__ w,   // (D,3)
                               float* __restrict__ out) {
    int row = blockIdx.x;            // b*S + s
    int s   = row % S;
    for (int d = threadIdx.x; d < D; d += blockDim.x) {
        const float* wd = w + (long long)d * 3;
        float acc = wd[1] * h[(long long)row * D + d];
        if (s > 0)     acc = fmaf(wd[0], h[(long long)(row - 1) * D + d], acc);
        if (s < S - 1) acc = fmaf(wd[2], h[(long long)(row + 1) * D + d], acc);
        out[(long long)row * D + d] = acc;
    }
}

// ---------------- 4. depthwise conv k=64, pad (32,31), LDS tiled ------
// tile: 64 channels x 32 outputs; input rows needed: 32+63 = 95
__global__ void dwconv64_kernel(const float* __restrict__ in,
                                const float* __restrict__ w,    // (D,64)
                                const float* __restrict__ res,  // may be null
                                float* __restrict__ out) {
    __shared__ float sIn[95 * 64];
    __shared__ float sW[64 * 64];    // [k][d] to avoid bank conflicts
    int d0 = blockIdx.x * 64;
    int s0 = blockIdx.y * 32;
    int b  = blockIdx.z;
    int tx = threadIdx.x & 63;       // channel within tile
    int tz = threadIdx.x >> 6;       // 0..3

#ifdef HAS_ASYNC_LDS
    for (int k = tz; k < 64; k += 4) {
        gas_ptr gp = (gas_ptr)(w + (long long)(d0 + tx) * 64 + k);
        las_ptr lp = (las_ptr)&sW[k * 64 + tx];
        __builtin_amdgcn_global_load_async_to_lds_b32(gp, lp, 0, 0);
    }
    for (int r = tz; r < 95; r += 4) {
        int s = s0 + r - 32;
        if (s >= 0 && s < S) {
            gas_ptr gp = (gas_ptr)(in + ((long long)b * S + s) * D + d0 + tx);
            las_ptr lp = (las_ptr)&sIn[r * 64 + tx];
            __builtin_amdgcn_global_load_async_to_lds_b32(gp, lp, 0, 0);
        } else {
            sIn[r * 64 + tx] = 0.f;
        }
    }
    wait_async_all();
#else
    for (int k = tz; k < 64; k += 4)
        sW[k * 64 + tx] = w[(long long)(d0 + tx) * 64 + k];
    for (int r = tz; r < 95; r += 4) {
        int s = s0 + r - 32;
        float v = (s >= 0 && s < S) ? in[((long long)b * S + s) * D + d0 + tx] : 0.f;
        sIn[r * 64 + tx] = v;
    }
#endif
    __syncthreads();
    for (int j = 0; j < 8; ++j) {
        int sl = tz * 8 + j;
        float acc = 0.f;
#pragma unroll
        for (int k = 0; k < 64; ++k)
            acc = fmaf(sW[k * 64 + tx], sIn[(sl + k) * 64 + tx], acc);
        long long o = ((long long)b * S + s0 + sl) * D + d0 + tx;
        if (res) acc += res[o];
        out[o] = acc;
    }
}

// ---------------- 5. silu(mix) -> bf16 A operand ----------------
__global__ void silumix_bf16_kernel(const float* __restrict__ g3,
                                    const float* __restrict__ mix,  // (D)
                                    unsigned short* __restrict__ ybf,
                                    long long n) {
    long long stride = (long long)gridDim.x * blockDim.x;
    for (long long i = (long long)blockIdx.x * blockDim.x + threadIdx.x; i < n; i += stride) {
        int d = (int)(i % D);
        float v = g3[i] * mix[d];
        float si = v / (1.f + __expf(-v));
        ybf[i] = f2bf(si);
    }
}

// ---------------- 6a. fp32 -> bf16 elementwise ----------------
__global__ void f2bf_kernel(const float* __restrict__ in,
                            unsigned short* __restrict__ out, long long n) {
    long long stride = (long long)gridDim.x * blockDim.x;
    for (long long i = (long long)blockIdx.x * blockDim.x + threadIdx.x; i < n; i += stride)
        out[i] = f2bf(in[i]);
}

// ---------------- 6b. pack transposed weight into WMMA-B fragment layout --
// Source: W[e*D + d] row-major (e=N col, d=K). B[k][n] = W[n*D + k].
// Dest:   Bp[((tile*KT) + kt)*512 + lane*16 + j], lane's 16 values are
//         k = kt*32 + (lane>>4)*16 + j  (contiguous in source row!)
__global__ void pack_Bt_kernel(const float* __restrict__ W,
                               unsigned short* __restrict__ Bp) {
    int idx  = blockIdx.x * blockDim.x + threadIdx.x;   // (tile, kt, lane)
    int lane = idx & 31;
    int kt   = (idx >> 5) % KT;
    int tile = idx / (32 * KT);
    int n     = tile * 16 + (lane & 15);
    int kbase = kt * 32 + ((lane >> 4) << 4);
    const float* src = W + (long long)n * D + kbase;
    unsigned short* dst = Bp + ((long long)tile * KT + kt) * 512 + lane * 16;
#pragma unroll
    for (int j = 0; j < 16; ++j) dst[j] = f2bf(src[j]);
}

// ---------------- 6c. pack K-major weight (out_W: K x N) into frag layout --
// grid: (N/256, K/32); coalesced reads, scattered u16 writes (runs once).
__global__ void pack_Bk_kernel(const float* __restrict__ W,
                               unsigned short* __restrict__ Bp, int N) {
    int n  = blockIdx.x * 256 + threadIdx.x;
    int k0 = blockIdx.y * 32;
    int tile = n >> 4;
    long long tbase = ((long long)tile * KT + (k0 >> 5)) * 512;
#pragma unroll 4
    for (int k = 0; k < 32; ++k) {
        float v = W[(long long)(k0 + k) * N + n];
        int lane = (n & 15) | (k & 16);      // k>=16 -> lanes 16..31
        int j    = k & 15;
        Bp[tbase + lane * 16 + j] = f2bf(v);
    }
}

// ---------------- 7. WMMA GEMM: C = A(MxK,bf16) * B + bias ----------------
// B pre-packed in fragment layout. One wave computes a 32(M) x 64(N) block:
// 2 M-tiles x 4 N-tiles = 8 wmma per K-step; A reused 4x, B reused 2x.
// The 8 waves of a block tile a 128x128 region (4 sm x 2 sn) so B panels are
// shared by 4 waves and A panels by 2 waves through WGP$/L2 locality.
__global__ void wmma_gemm_kernel(const unsigned short* __restrict__ A,
                                 const unsigned short* __restrict__ Bp,
                                 const float* __restrict__ bias,   // may be null
                                 float* __restrict__ C,
                                 int M, int N, int K) {
    int lane = threadIdx.x & 31;
    int w    = threadIdx.x >> 5;                // 0..7
    int regionsN = N >> 7;                      // 128-wide regions
    int region   = blockIdx.x;
    if (region >= (M >> 7) * regionsN) return;  // wave-uniform: EXEC stays all-1s
    int rm = region / regionsN;
    int rn = region - rm * regionsN;
    int sm = rm * 4 + (w >> 1);                 // 32-row supertile index
    int sn = rn * 2 + (w & 1);                  // 64-col supertile index

    int m    = lane & 15;
    int half = lane >> 4;
    int kTiles = K >> 5;

    const unsigned short* a0 = A + (long long)(sm * 32 + m) * K;
    const unsigned short* a1 = a0 + (long long)16 * K;
    // per-lane packed-B stream for each of the 4 N-tiles
    const unsigned short* b0 = Bp + ((long long)(sn * 4) * kTiles) * 512 + lane * 16;

    v8f acc[8];
#pragma unroll
    for (int i = 0; i < 8; ++i)
        acc[i] = (v8f){0.f, 0.f, 0.f, 0.f, 0.f, 0.f, 0.f, 0.f};

    for (int kt = 0; kt < kTiles; ++kt) {
        int k0 = kt << 5;
        FragA fa0, fa1;
        const unsigned int* p0 = (const unsigned int*)(a0 + k0);
        const unsigned int* p1 = (const unsigned int*)(a1 + k0);
        __builtin_prefetch(p0 + 16, 0, 1);
        __builtin_prefetch(p1 + 16, 0, 1);
#pragma unroll
        for (int vI = 0; vI < 4; ++vI) {
            fa0.u[vI]     = p0[vI + 8 * half];
            fa0.u[4 + vI] = p0[4 + vI + 8 * half];
            fa1.u[vI]     = p1[vI + 8 * half];
            fa1.u[4 + vI] = p1[4 + vI + 8 * half];
        }
#pragma unroll
        for (int t = 0; t < 4; ++t) {
            const unsigned int* bp =
                (const unsigned int*)(b0 + ((long long)t * kTiles + kt) * 512);
            FragB fb;
#pragma unroll
            for (int vI = 0; vI < 8; ++vI) fb.u[vI] = bp[vI];   // 32B contiguous
            acc[t] = __builtin_amdgcn_wmma_f32_16x16x32_bf16(
                false, fa0.v, false, fb.v, (short)0, acc[t], false, false);
            acc[4 + t] = __builtin_amdgcn_wmma_f32_16x16x32_bf16(
                false, fa1.v, false, fb.v, (short)0, acc[4 + t], false, false);
        }
    }

    // C/D layout: VGPR r -> M = r + 8*half, N = lane&15
    // Non-temporal stores: the C stream must not evict L2-resident B panels.
#pragma unroll
    for (int t = 0; t < 4; ++t) {
        int nn = (sn * 64 + t * 16) + (lane & 15);
        float bv = bias ? bias[nn] : 0.f;
#pragma unroll
        for (int r = 0; r < 8; ++r) {
            int mm0 = sm * 32 + r + 8 * half;
            __builtin_nontemporal_store(acc[t][r] + bv,
                                        &C[(long long)mm0 * N + nn]);
            __builtin_nontemporal_store(acc[4 + t][r] + bv,
                                        &C[(long long)(mm0 + 16) * N + nn]);
        }
    }
}

// ---------------- 8. gated residual combine ----------------
__global__ void combine_kernel(float* __restrict__ x,
                               const float* __restrict__ refined,
                               const float* __restrict__ gatepre,
                               long long n) {
    long long stride = (long long)gridDim.x * blockDim.x;
    for (long long i = (long long)blockIdx.x * blockDim.x + threadIdx.x; i < n; i += stride) {
        float sg = 1.f / (1.f + __expf(-gatepre[i]));
        x[i] = sg * refined[i] + (1.f - sg) * x[i];
    }
}

// ---------------- launch ----------------
extern "C" void kernel_launch(void* const* d_in, const int* in_sizes, int n_in,
                              void* d_out, int out_size, void* d_ws, size_t ws_size,
                              hipStream_t stream) {
    const int*   ids     = (const int*)d_in[0];
    const float* emb     = (const float*)d_in[1];
    const float* local_w = (const float*)d_in[2];   // (NL,D,3)
    const float* filt_w  = (const float*)d_in[3];   // (NL,2,D,64)
    const float* mix_w   = (const float*)d_in[4];   // (NL,D)
    const float* ln1_g   = (const float*)d_in[5];
    const float* ln1_b   = (const float*)d_in[6];
    const float* ref_W   = (const float*)d_in[7];   // (NL,D,D)
    const float* ref_b   = (const float*)d_in[8];
    const float* gate_W  = (const float*)d_in[9];
    const float* gate_b  = (const float*)d_in[10];
    const float* lnf_g   = (const float*)d_in[11];
    const float* lnf_b   = (const float*)d_in[12];
    const float* out_W   = (const float*)d_in[13];  // (D,V)
    const float* out_b   = (const float*)d_in[14];
    float* logits = (float*)d_out;

    // ---- workspace carve ----
    char* ws = (char*)d_ws;
    auto carve = [&](size_t bytes) -> char* {
        char* p = ws;
        ws += (bytes + 255) & ~(size_t)255;
        return p;
    };
    float* xcur    = (float*)carve(BSD * sizeof(float));
    float* hbuf    = (float*)carve(BSD * sizeof(float));
    float* g1      = (float*)carve(BSD * sizeof(float));
    float* g2      = (float*)carve(BSD * sizeof(float));
    float* refined = (float*)carve(BSD * sizeof(float));
    float* gatepre = (float*)carve(BSD * sizeof(float));
    unsigned short* ybf    = (unsigned short*)carve(BSD * sizeof(unsigned short));
    unsigned short* xbf    = (unsigned short*)carve(BSD * sizeof(unsigned short));
    unsigned short* refWp  = (unsigned short*)carve((size_t)D * D * sizeof(unsigned short));
    unsigned short* gateWp = (unsigned short*)carve((size_t)D * D * sizeof(unsigned short));
    unsigned short* outWp  = (unsigned short*)carve((size_t)D * V * sizeof(unsigned short));

    dim3 blk(256);

    // embedding + positional
    embed_kernel<<<dim3(BS), blk, 0, stream>>>(ids, emb, xcur);

    const dim3 convGrid(D / 64, S / 32, B);
    const int packT_blocks = (D / 16) * KT * 32 / 256;        // 256
    const int gemmDD_blocks = (BS / 128) * (D / 128);         // 32*8   = 256
    const int gemmV_blocks  = (BS / 128) * (V / 128);         // 32*250 = 8000

    for (int l = 0; l < NL; ++l) {
        ln_kernel<<<dim3(BS), blk, 0, stream>>>(xcur, ln1_g + (long long)l * D,
                                                ln1_b + (long long)l * D, hbuf);
        dwconv3_kernel<<<dim3(BS), blk, 0, stream>>>(hbuf, local_w + (long long)l * D * 3, g1);
        dwconv64_kernel<<<convGrid, blk, 0, stream>>>(
            g1, filt_w + ((long long)l * 2 + 0) * D * 64, hbuf, g2);
        dwconv64_kernel<<<convGrid, blk, 0, stream>>>(
            g2, filt_w + ((long long)l * 2 + 1) * D * 64, nullptr, g1);
        silumix_bf16_kernel<<<dim3(2048), blk, 0, stream>>>(g1, mix_w + (long long)l * D, ybf, BSD);
        f2bf_kernel<<<dim3(2048), blk, 0, stream>>>(xcur, xbf, BSD);
        // pack weights into WMMA-B fragment layout (B[k][n] = W[n*D+k])
        pack_Bt_kernel<<<dim3(packT_blocks), blk, 0, stream>>>(ref_W + (long long)l * D * D, refWp);
        pack_Bt_kernel<<<dim3(packT_blocks), blk, 0, stream>>>(gate_W + (long long)l * D * D, gateWp);
        // refined = y @ ref_W^T + ref_b
        wmma_gemm_kernel<<<dim3(gemmDD_blocks), blk, 0, stream>>>(
            ybf, refWp, ref_b + (long long)l * D, refined, BS, D, D);
        // gatepre = x @ gate_W^T + gate_b
        wmma_gemm_kernel<<<dim3(gemmDD_blocks), blk, 0, stream>>>(
            xbf, gateWp, gate_b + (long long)l * D, gatepre, BS, D, D);
        combine_kernel<<<dim3(2048), blk, 0, stream>>>(xcur, refined, gatepre, BSD);
    }

    // final LN -> bf16
    ln_kernel<<<dim3(BS), blk, 0, stream>>>(xcur, lnf_g, lnf_b, hbuf);
    f2bf_kernel<<<dim3(2048), blk, 0, stream>>>(hbuf, xbf, BSD);
    // pack out_W (K-major D x V) into fragment layout
    pack_Bk_kernel<<<dim3(V / 256, D / 32), blk, 0, stream>>>(out_W, outWp, V);
    // logits = x @ out_W + out_b
    wmma_gemm_kernel<<<dim3(gemmV_blocks), blk, 0, stream>>>(
        xbf, outWp, out_b, logits, BS, V, D);
}
